// FSpectralConv2d_47579647705091
// MI455X (gfx1250) — matrix operand order repaired and checked by reading, not compiled, not run on hardware
//
#include <hip/hip_runtime.h>

// ---------------------------------------------------------------------------
// FNO spectral block for MI455X (gfx1250, wave32, WMMA 16x16x32 bf16)
// Spectral conv = GEMM chains vs fixed DFT matrices; FFN fused with LDS.
// All transposes go through padded LDS (conflict-free), all global traffic
// is lane-coalesced or 32B-contiguous per lane.
// ---------------------------------------------------------------------------

typedef __attribute__((ext_vector_type(16))) __bf16 v16bf;
typedef __attribute__((ext_vector_type(8)))  float  v8f;

union Frag { v16bf v; uint4 q[2]; };

static __device__ __forceinline__ unsigned short f2bfu(float f) {
  unsigned u = __builtin_bit_cast(unsigned, f);
  u += 0x7FFFu + ((u >> 16) & 1u);          // round-to-nearest-even
  return (unsigned short)(u >> 16);
}

// A-matrix fragment (16x32 bf16): row = row0 + (lane&15);
// lane<16 -> K k0..k0+15, lane>=16 -> K k0+16..k0+31.
static __device__ __forceinline__ v16bf frag_a(const unsigned short* __restrict__ base,
                                               int row0, int ld, int k0, int lane) {
  const int r  = row0 + (lane & 15);
  const int hi = (lane >> 4) * 16;
  Frag f;
  f.q[0] = *(const uint4*)(base + (size_t)r * ld + k0 + hi);
  f.q[1] = *(const uint4*)(base + (size_t)r * ld + k0 + 8 + hi);
  return f.v;
}

// B-matrix fragment (32x16 bf16) from a [col][K] buffer:
// lane<16 -> col, K k0..k0+15 ; lane>=16 -> col, K k0+16..k0+31.
static __device__ __forceinline__ v16bf frag_b(const unsigned short* __restrict__ base,
                                               int col0, int ld, int k0, int lane) {
  const int c = col0 + (lane & 15);
  const unsigned short* p = base + (size_t)c * ld + k0 + ((lane >> 4) << 4);
  Frag f;
  f.q[0] = *(const uint4*)(p);
  f.q[1] = *(const uint4*)(p + 8);
  return f.v;
}

static __device__ __forceinline__ v8f wmma_bf16(v16bf a, v16bf b, v8f c) {
  return __builtin_amdgcn_wmma_f32_16x16x32_bf16(false, a, false, b, (short)0, c,
                                                 false, false);
}

// ---------------- geometry ----------------
#define BB    16
#define MM    128
#define NN    128
#define DD    128
#define MODES 32
#define HH    512
#define KC    64            // 32 re + 32 im spectral rows
#define BM    (BB * MM)     // 2048
#define SS    (BB * MM * NN)
#define LZP   72            // [o][k2] LDS pitch (144B: aligned, bank step 4)
#define LAP   136           // FFN A pitch  (272B)
#define LHP   520           // FFN H pitch  (1040B)
#define LTP   130           // transpose LDS pitch (odd word step -> no conflicts)

// ===========================================================================
// prep: DFT matrices.  Ft[64][128] forward (ortho), It[128][64] inverse with
// conjugate-symmetry 2x factors folded in.
// ===========================================================================
__global__ void k_prep_dft(unsigned short* __restrict__ Ft,
                           unsigned short* __restrict__ It) {
  const int gid = blockIdx.x * blockDim.x + threadIdx.x;   // 0..8191
  const float s = 0.08838834764831845f;                    // 1/sqrt(128)
  {
    const int kp = gid >> 7, n = gid & 127, k = kp & 31;
    const float ang = 6.283185307179586f * (float)(k * n) / 128.0f;
    Ft[gid] = f2bfu(((kp < 32) ? __cosf(ang) : -__sinf(ang)) * s);
  }
  {
    const int n = gid >> 6, k2 = gid & 63, k = k2 & 31;
    const float ck = (k == 0) ? 1.0f : 2.0f;
    const float ang = 6.283185307179586f * (float)(k * n) / 128.0f;
    It[gid] = f2bfu(((k2 < 32) ? ck * __cosf(ang) : -ck * __sinf(ang)) * s);
  }
}

// fw [i][o][k][2] fp32  ->  Wre/Wim [k][o][i] bf16 (B operand: K=i contiguous)
__global__ void k_prep_fw(const float* __restrict__ fw,
                          unsigned short* __restrict__ Wre,
                          unsigned short* __restrict__ Wim) {
  const int gid = blockIdx.x * blockDim.x + threadIdx.x;   // 0..524287
  const int i = gid & 127, o = (gid >> 7) & 127, k = gid >> 14;
  const size_t src = (((size_t)i * 128 + o) * MODES + k) * 2;
  Wre[gid] = f2bfu(fw[src]);
  Wim[gid] = f2bfu(fw[src + 1]);
}

// w1 [128][512] -> w1t [512][128] bf16 ; w2 [512][128] -> w2t [128][512] bf16
__global__ void k_prep_ffnw(const float* __restrict__ w1, const float* __restrict__ w2,
                            unsigned short* __restrict__ w1t,
                            unsigned short* __restrict__ w2t) {
  const int gid = blockIdx.x * blockDim.x + threadIdx.x;   // 0..65535
  { const int h = gid >> 7, d = gid & 127;  w1t[gid] = f2bfu(w1[(size_t)d * HH + h]); }
  { const int o = gid >> 9, h = gid & 511;  w2t[gid] = f2bfu(w2[(size_t)h * DD + o]); }
}

// ===========================================================================
// tiled transpose: one 128x128 plane per block, through padded LDS.
// xaxis=0: x[b][m][:][:] (r=n)  -> xbY[(b,m)][i][n]
// xaxis=1: x[b][:][n][:] (r=m)  -> xbX[(b,n)][i][m]
// Coalesced fp32 reads, coalesced packed-u32 bf16 writes.
// ===========================================================================
__global__ void k_tr(const float* __restrict__ x, unsigned short* __restrict__ out,
                     int xaxis) {
  __shared__ unsigned short lds[128 * LTP];
  const int blk = blockIdx.x;                // 2048 planes
  size_t base; int rstride;
  if (xaxis) { base = (size_t)(blk >> 7) * 2097152 + (size_t)(blk & 127) * 128;
               rstride = 16384; }
  else       { base = (size_t)blk * 16384;  rstride = 128; }
  for (int e = threadIdx.x; e < 128 * 128; e += blockDim.x) {
    const int r = e >> 7, i = e & 127;       // consecutive threads -> i: coalesced
    lds[i * LTP + r] = f2bfu(x[base + (size_t)r * rstride + i]);
  }
  __syncthreads();
  const size_t outbase = (size_t)blk * 16384;
  unsigned* outw = (unsigned*)(out + outbase);
  for (int e = threadIdx.x; e < 128 * 64; e += blockDim.x) {
    const int i = e >> 6, r2 = e & 63;       // consecutive threads -> r2: coalesced
    const unsigned lo = lds[i * LTP + 2 * r2], hiw = lds[i * LTP + 2 * r2 + 1];
    outw[i * 64 + r2] = lo | (hiw << 16);
  }
}

// ===========================================================================
// stage 1: truncated forward DFT as GEMM.  Y[k'][bm][i] = Ft[k'][:] . xT[bm][i][:]
// ===========================================================================
__global__ void k_dft(const unsigned short* __restrict__ Ft,   // [64][128]
                      const unsigned short* __restrict__ xT,   // [BM][128][128]
                      unsigned short* __restrict__ Y) {        // [64][BM][128]
  const int bm   = blockIdx.x;
  const int lane = threadIdx.x & 31;
  const int wave = threadIdx.x >> 5;
  const int row0 = wave * 16;
  const unsigned short* Bb = xT + (size_t)bm * (128 * 128);
  v8f acc[8] = {};
#pragma unroll
  for (int k0 = 0; k0 < 128; k0 += 32) {
    const v16bf a = frag_a(Ft, row0, 128, k0, lane);
#pragma unroll
    for (int t = 0; t < 8; ++t) {
      const v16bf b = frag_b(Bb, t * 16, 128, k0, lane);
      acc[t] = wmma_bf16(a, b, acc[t]);
    }
  }
  const int cc = lane & 15, hi = (lane >> 4) * 8;
#pragma unroll
  for (int t = 0; t < 8; ++t)
#pragma unroll
    for (int r = 0; r < 8; ++r)
      Y[(size_t)(row0 + r + hi) * (BM * 128) + (size_t)bm * 128 + t * 16 + cc] =
          f2bfu(acc[t][r]);
}

// ===========================================================================
// stage 2: per-mode complex channel mix.  Z[k2][bm][o]  (k2: re=k, im=k+32)
// K-outer loop so the shared Yre/Yim A-fragments feed both o-tiles.
// Stores are lane-coalesced along o.
// ===========================================================================
__global__ void k_mix(const unsigned short* __restrict__ Y,    // [64][BM][128]
                      const unsigned short* __restrict__ Wre,  // [32][128][128]
                      const unsigned short* __restrict__ Wim,
                      unsigned short* __restrict__ Z) {        // [64][BM][128]
  const int k    = blockIdx.y;
  const int bm0  = blockIdx.x * 16;
  const int lane = threadIdx.x & 31;
  const int wave = threadIdx.x >> 5;
  const unsigned short* Are = Y + (size_t)k * (BM * 128);
  const unsigned short* Aim = Y + (size_t)(k + 32) * (BM * 128);
  const unsigned short* Bre = Wre + (size_t)k * (128 * 128);
  const unsigned short* Bim = Wim + (size_t)k * (128 * 128);
  __builtin_prefetch(Bre + (size_t)(wave * 32) * 128, 0, 1);
  __builtin_prefetch(Bim + (size_t)(wave * 32) * 128, 0, 1);
  v8f rr[2] = {}, ii[2] = {}, ri[2] = {}, ir[2] = {};
#pragma unroll
  for (int k0 = 0; k0 < 128; k0 += 32) {
    const v16bf are = frag_a(Are, bm0, 128, k0, lane);
    const v16bf aim = frag_a(Aim, bm0, 128, k0, lane);
#pragma unroll
    for (int ot = 0; ot < 2; ++ot) {
      const int o0 = wave * 32 + ot * 16;
      const v16bf bre = frag_b(Bre, o0, 128, k0, lane);
      const v16bf bim = frag_b(Bim, o0, 128, k0, lane);
      rr[ot] = wmma_bf16(are, bre, rr[ot]);
      ii[ot] = wmma_bf16(aim, bim, ii[ot]);
      ri[ot] = wmma_bf16(are, bim, ri[ot]);
      ir[ot] = wmma_bf16(aim, bre, ir[ot]);
    }
  }
  const int cc = lane & 15, hi = (lane >> 4) * 8;
#pragma unroll
  for (int ot = 0; ot < 2; ++ot) {
    const int o0 = wave * 32 + ot * 16;
#pragma unroll
    for (int r = 0; r < 8; ++r) {
      const int bm = bm0 + r + hi, o = o0 + cc;
      Z[((size_t)k * BM + bm) * 128 + o]        = f2bfu(rr[ot][r] - ii[ot][r]);
      Z[((size_t)(k + 32) * BM + bm) * 128 + o] = f2bfu(ri[ot][r] + ir[ot][r]);
    }
  }
}

// ===========================================================================
// stage 3: inverse DFT as GEMM, fp32 accumulate into h[b][m][n][o].
// Z slice [64 k2][128 o] (row stride BM*128) is staged into LDS transposed
// to [o][k2] (pitch 72 -> conflict-free b128 fragment reads).
// ===========================================================================
__global__ void k_idft(const unsigned short* __restrict__ It,  // [128][64]
                       const unsigned short* __restrict__ Z,   // [64][BM][128]
                       float* __restrict__ h, int xaxis, int accumulate) {
  __shared__ unsigned short ldsZ[128 * LZP];
  const int bm   = blockIdx.x;
  const int lane = threadIdx.x & 31;
  const int wave = threadIdx.x >> 5;
  // cooperative coalesced load + transpose of Z[:, bm, :]
  for (int u = threadIdx.x; u < 64 * 64; u += blockDim.x) {
    const int k2 = u >> 6, o2 = u & 63;      // consecutive threads -> o2
    const unsigned v =
        *(const unsigned*)(Z + ((size_t)k2 * BM + bm) * 128 + o2 * 2);
    ldsZ[(2 * o2) * LZP + k2]     = (unsigned short)v;
    ldsZ[(2 * o2 + 1) * LZP + k2] = (unsigned short)(v >> 16);
  }
  __syncthreads();
  size_t base; int mul;
  if (xaxis) { base = (size_t)(bm >> 7) * 16384 + (bm & 127); mul = 128; }
  else       { base = (size_t)bm * 128;                        mul = 1;  }
#pragma unroll
  for (int rs = 0; rs < 2; ++rs) {
    const int n0 = wave * 16 + rs * 64;
    v8f acc[8] = {};
#pragma unroll
    for (int k0 = 0; k0 < KC; k0 += 32) {
      const v16bf a = frag_a(It, n0, KC, k0, lane);
#pragma unroll
      for (int t = 0; t < 8; ++t) {
        const v16bf b = frag_b(ldsZ, t * 16, LZP, k0, lane);
        acc[t] = wmma_bf16(a, b, acc[t]);
      }
    }
    const int cc = lane & 15, hi = (lane >> 4) * 8;
#pragma unroll
    for (int t = 0; t < 8; ++t)
#pragma unroll
      for (int r = 0; r < 8; ++r) {
        const size_t idx = (base + (size_t)(n0 + r + hi) * mul) * 128 + t * 16 + cc;
        float v = acc[t][r];
        if (accumulate) v += h[idx];
        h[idx] = v;
      }
  }
}

// ===========================================================================
// fused FFN: relu(h@w1 + b1) @ w2 + b2.  64 rows/block, hidden kept in LDS.
// Padded pitches (136/520) keep all ds_load_b128 fragment reads conflict-free.
// ===========================================================================
__global__ void k_ffn(const float* __restrict__ h,
                      const unsigned short* __restrict__ w1t,  // [512][128]
                      const float* __restrict__ b1,
                      const unsigned short* __restrict__ w2t,  // [128][512]
                      const float* __restrict__ b2,
                      float* __restrict__ out) {
  extern __shared__ unsigned short smem[];
  unsigned short* ldsA = smem;                  // [64][LAP]
  unsigned short* ldsH = smem + 64 * LAP;       // [64][LHP]
  const int s0   = blockIdx.x * 64;
  const int lane = threadIdx.x & 31;
  const int wave = threadIdx.x >> 5;
  for (int idx = threadIdx.x; idx < 64 * 128; idx += blockDim.x)
    ldsA[(idx >> 7) * LAP + (idx & 127)] = f2bfu(h[(size_t)s0 * 128 + idx]);
  __syncthreads();
  const int cc = lane & 15, hi = (lane >> 4) * 8;
  // GEMM1: 16-row strip per wave x 512 hidden, bias+relu, into LDS
  for (int t = 0; t < 32; ++t) {
    __builtin_prefetch(w1t + (size_t)(t * 16) * 128, 0, 1);
    v8f acc = {};
#pragma unroll
    for (int k0 = 0; k0 < 128; k0 += 32) {
      const v16bf a = frag_a(ldsA, wave * 16, LAP, k0, lane);
      const v16bf b = frag_b(w1t, t * 16, 128, k0, lane);
      acc = wmma_bf16(a, b, acc);
    }
    const int col = t * 16 + cc;
    const float bias = b1[col];
#pragma unroll
    for (int r = 0; r < 8; ++r) {
      float v = acc[r] + bias;
      v = v > 0.0f ? v : 0.0f;
      ldsH[(size_t)(wave * 16 + r + hi) * LHP + col] = f2bfu(v);
    }
  }
  __syncthreads();
  // GEMM2: 16-row strip per wave x 128 out, K=512
  for (int t = 0; t < 8; ++t) {
    v8f acc = {};
#pragma unroll
    for (int k0 = 0; k0 < HH; k0 += 32) {
      const v16bf a = frag_a(ldsH, wave * 16, LHP, k0, lane);
      const v16bf b = frag_b(w2t, t * 16, HH, k0, lane);
      acc = wmma_bf16(a, b, acc);
    }
    const int col = t * 16 + cc;
    const float bias = b2[col];
#pragma unroll
    for (int r = 0; r < 8; ++r)
      out[(size_t)(s0 + wave * 16 + r + hi) * 128 + col] = acc[r] + bias;
  }
}

// ===========================================================================
extern "C" void kernel_launch(void* const* d_in, const int* in_sizes, int n_in,
                              void* d_out, int out_size, void* d_ws, size_t ws_size,
                              hipStream_t stream) {
  (void)in_sizes; (void)n_in; (void)out_size; (void)ws_size;
  const float* x   = (const float*)d_in[0];
  const float* fw0 = (const float*)d_in[1];
  const float* fw1 = (const float*)d_in[2];
  const float* w1  = (const float*)d_in[3];
  const float* b1  = (const float*)d_in[4];
  const float* w2  = (const float*)d_in[5];
  const float* b2  = (const float*)d_in[6];
  float* out = (float*)d_out;

  char* p = (char*)d_ws;
  auto alloc = [&](size_t bytes) -> void* {
    void* r = (void*)p;
    p += (bytes + 255) & ~(size_t)255;
    return r;
  };
  unsigned short* Ft   = (unsigned short*)alloc(64 * 128 * 2);
  unsigned short* It   = (unsigned short*)alloc(128 * 64 * 2);
  unsigned short* W0re = (unsigned short*)alloc((size_t)MODES * 128 * 128 * 2);
  unsigned short* W0im = (unsigned short*)alloc((size_t)MODES * 128 * 128 * 2);
  unsigned short* W1re = (unsigned short*)alloc((size_t)MODES * 128 * 128 * 2);
  unsigned short* W1im = (unsigned short*)alloc((size_t)MODES * 128 * 128 * 2);
  unsigned short* w1t  = (unsigned short*)alloc((size_t)HH * DD * 2);
  unsigned short* w2t  = (unsigned short*)alloc((size_t)DD * HH * 2);
  unsigned short* xbY  = (unsigned short*)alloc((size_t)33554432 * 2);
  unsigned short* xbX  = (unsigned short*)alloc((size_t)33554432 * 2);
  unsigned short* Ybuf = (unsigned short*)alloc((size_t)KC * BM * 128 * 2);
  unsigned short* Zbuf = (unsigned short*)alloc((size_t)KC * BM * 128 * 2);
  float*          hbuf = (float*)alloc((size_t)33554432 * 4);

  // ---- prep ----
  k_prep_dft<<<64, 128, 0, stream>>>(Ft, It);
  k_prep_fw<<<2048, 256, 0, stream>>>(fw0, W0re, W0im);
  k_prep_fw<<<2048, 256, 0, stream>>>(fw1, W1re, W1im);
  k_prep_ffnw<<<256, 256, 0, stream>>>(w1, w2, w1t, w2t);
  k_tr<<<2048, 256, 0, stream>>>(x, xbY, 0);
  k_tr<<<2048, 256, 0, stream>>>(x, xbX, 1);

  // ---- y-axis spectral conv ----
  k_dft<<<BM, 128, 0, stream>>>(Ft, xbY, Ybuf);
  k_mix<<<dim3(BM / 16, MODES), 128, 0, stream>>>(Ybuf, W0re, W0im, Zbuf);
  k_idft<<<BM, 128, 0, stream>>>(It, Zbuf, hbuf, 0, 0);

  // ---- x-axis spectral conv (accumulate) ----
  k_dft<<<BM, 128, 0, stream>>>(Ft, xbX, Ybuf);
  k_mix<<<dim3(BM / 16, MODES), 128, 0, stream>>>(Ybuf, W1re, W1im, Zbuf);
  k_idft<<<BM, 128, 0, stream>>>(It, Zbuf, hbuf, 1, 1);

  // ---- fused FFN ----
  const size_t ldsBytes = (size_t)(64 * LAP + 64 * LHP) * sizeof(unsigned short);
  hipFuncSetAttribute((const void*)k_ffn,
                      hipFuncAttributeMaxDynamicSharedMemorySize, (int)ldsBytes);
  k_ffn<<<SS / 64, 128, ldsBytes, stream>>>(hbuf, w1t, b1, w2t, b2, out);
}
